// MultiHeadAttention_2894807958021
// MI455X (gfx1250) — matrix operand order repaired and checked by reading, not compiled, not run on hardware
//
#include <hip/hip_runtime.h>

// ---------------------------------------------------------------------------
// MI455X / gfx1250 fused dual-softmax multi-head attention.
// B=4, S=2048, D=512, H=8, DK=64.  wave32, WMMA f16 -> f32.
// GEMMs stage tiles in LDS via GLOBAL_LOAD_ASYNC_TO_LDS_B128 (ASYNCcnt).
// ---------------------------------------------------------------------------

#define Bc 4
#define Sc 2048
#define Dc 512
#define Hc 8
#define DKc 64
#define SCALE 0.125f   // 1/sqrt(64)

typedef _Float16 v8h  __attribute__((ext_vector_type(8)));
typedef _Float16 v16h __attribute__((ext_vector_type(16)));
typedef float    v8f  __attribute__((ext_vector_type(8)));

#define WMMA_F16(a, b, c) \
  __builtin_amdgcn_wmma_f32_16x16x32_f16(false, (a), false, (b), (short)0, (c), false, false)

__device__ __forceinline__ v8f vzero8() {
  v8f z;
#pragma unroll
  for (int i = 0; i < 8; ++i) z[i] = 0.0f;
  return z;
}

// ---- CDNA5 async copy: global -> LDS, tracked by ASYNCcnt ------------------
// the vdst VGPR gives the LDS byte address; LDS offset of a __shared__
// object is the low 32 bits of its generic address (LDS aperture rule).
__device__ __forceinline__ unsigned lds_off(const void* p) {
  return (unsigned)(unsigned long long)p;
}
__device__ __forceinline__ void async_load_b128(unsigned lds_byte_off,
                                                const void* gaddr) {
  asm volatile("global_load_async_to_lds_b128 %0, %1, off"
               :: "v"(lds_byte_off), "v"(gaddr) : "memory");
}
__device__ __forceinline__ void wait_async0() {
  asm volatile("s_wait_asynccnt 0" ::: "memory");
}

// A fragment: 16x32 f16, row-major source with leading dim `ld` (halves).
// Lane (m = lane&15, hg = lane>>4) holds K = hg*16 + 0..15 (contiguous).
__device__ __forceinline__ v16h load_a_frag(const _Float16* __restrict__ X,
                                            int ld, int row0, int k0, int lane) {
  const int hg = lane >> 4, m = lane & 15;
  const _Float16* p = X + (size_t)(row0 + m) * ld + k0 + hg * 16;
  v8h lo = *(const v8h*)p;
  v8h hi = *(const v8h*)(p + 8);
  v16h a;
#pragma unroll
  for (int i = 0; i < 8; ++i) { a[i] = lo[i]; a[i + 8] = hi[i]; }
  return a;
}

// B fragment: logical B[k][n] = W[n][k] with W row-major, leading dim `ld`.
// Lane (n = lane&15, hg = lane>>4) holds K = hg*16 + 0..15 (contiguous).
__device__ __forceinline__ v16h load_b_frag(const _Float16* __restrict__ W,
                                            int ld, int n0, int k0, int lane) {
  const int hg = lane >> 4, n = lane & 15;
  return *(const v16h*)(W + (size_t)(n0 + n) * ld + k0 + hg * 16);
}

// ---------------------------------------------------------------------------
// Weight convert f32 -> f16
// ---------------------------------------------------------------------------
__global__ void f32_to_f16_kernel(const float* __restrict__ src,
                                  _Float16* __restrict__ dst, int n) {
  int i = blockIdx.x * blockDim.x + threadIdx.x;
  if (i < n) dst[i] = (_Float16)src[i];
}

// ---------------------------------------------------------------------------
// LayerNorm over D=512, one row per 256-thread block, f16 output.
// ---------------------------------------------------------------------------
__global__ void ln_f16_kernel(const float* __restrict__ x,
                              const float* __restrict__ gamma,
                              const float* __restrict__ beta,
                              _Float16* __restrict__ y) {
  __shared__ float red[256];
  const int row = blockIdx.x, tid = threadIdx.x;
  const float* xr = x + (size_t)row * Dc;
  float v0 = xr[tid], v1 = xr[tid + 256];
  red[tid] = v0 + v1;
  __syncthreads();
  for (int off = 128; off > 0; off >>= 1) {
    if (tid < off) red[tid] += red[tid + off];
    __syncthreads();
  }
  float mu = red[0] * (1.0f / Dc);
  __syncthreads();
  float d0 = v0 - mu, d1 = v1 - mu;
  red[tid] = d0 * d0 + d1 * d1;
  __syncthreads();
  for (int off = 128; off > 0; off >>= 1) {
    if (tid < off) red[tid] += red[tid + off];
    __syncthreads();
  }
  float rs = rsqrtf(red[0] * (1.0f / Dc) + 1e-6f);
  _Float16* yr = y + (size_t)row * Dc;
  yr[tid]       = (_Float16)(d0 * rs * gamma[tid] + beta[tid]);
  yr[tid + 256] = (_Float16)(d1 * rs * gamma[tid + 256] + beta[tid + 256]);
}

// ---------------------------------------------------------------------------
// WMMA GEMM  Y = X(MxK,f16) @ W(NxK,f16)^T  (i.e. nn.Linear).
// Block = 256 threads = 8 waves (2 M x 4 N); wave tile 64x32; block tile 128x128.
// Both 128x32 input tiles are staged in LDS via async copies (ASYNCcnt), then
// WMMA fragments are read back with ds_load_b128.
// mode 0: f16 store row-major          mode 1: f16 store V-transposed per batch
// mode 2: sigmoid(v + bias) f16 store  mode 3: f32 store (v + bias)
// ---------------------------------------------------------------------------
__global__ void gemm_ws_kernel(const _Float16* __restrict__ X,
                               const _Float16* __restrict__ W,
                               const float* __restrict__ bias,
                               void* __restrict__ Yout,
                               int M, int N, int K, int mode) {
  __shared__ __align__(16) _Float16 Xtile[128][32];
  __shared__ __align__(16) _Float16 Wtile[128][32];

  const int tid = threadIdx.x;
  const int lane = tid & 31, wid = tid >> 5;
  const int wm = wid & 1, wn = wid >> 1;
  const int rowBlk = blockIdx.x * 128;
  const int colBlk = blockIdx.y * 128;
  const int hg = lane >> 4, nl = lane & 15;

  // async-copy assignment: thread -> (row = tid>>2 and +64, 16B chunk = tid&3)
  const int crow = tid >> 2;               // 0..63
  const int cchk = (tid & 3) * 8;          // half-offset 0,8,16,24
  const unsigned ldsX0 = lds_off(&Xtile[crow][cchk]);
  const unsigned ldsX1 = lds_off(&Xtile[crow + 64][cchk]);
  const unsigned ldsW0 = lds_off(&Wtile[crow][cchk]);
  const unsigned ldsW1 = lds_off(&Wtile[crow + 64][cchk]);
  const _Float16* gX0 = X + (size_t)(rowBlk + crow) * K + cchk;
  const _Float16* gX1 = X + (size_t)(rowBlk + crow + 64) * K + cchk;
  const _Float16* gW0 = W + (size_t)(colBlk + crow) * K + cchk;
  const _Float16* gW1 = W + (size_t)(colBlk + crow + 64) * K + cchk;

  v8f acc[4][2];
#pragma unroll
  for (int i = 0; i < 4; ++i)
#pragma unroll
    for (int j = 0; j < 2; ++j) acc[i][j] = vzero8();

  for (int k0 = 0; k0 < K; k0 += 32) {
    // ---- stage 128x32 X and W tiles in LDS (async, 4 x b128 per thread) ----
    async_load_b128(ldsX0, gX0 + k0);
    async_load_b128(ldsX1, gX1 + k0);
    async_load_b128(ldsW0, gW0 + k0);
    async_load_b128(ldsW1, gW1 + k0);
    wait_async0();
    __syncthreads();

    // ---- fragments from LDS + 8 WMMA ----
    v16h a[4], b[2];
#pragma unroll
    for (int i = 0; i < 4; ++i) {
      const _Float16* xr = &Xtile[wm * 64 + i * 16 + nl][0];
      v8h lo = *(const v8h*)(xr + hg * 16);
      v8h hi = *(const v8h*)(xr + hg * 16 + 8);
#pragma unroll
      for (int e = 0; e < 8; ++e) { a[i][e] = lo[e]; a[i][e + 8] = hi[e]; }
    }
#pragma unroll
    for (int j = 0; j < 2; ++j) {
      const _Float16* wr = &Wtile[wn * 32 + j * 16 + nl][0];
      b[j] = *(const v16h*)(wr + hg * 16);
    }
#pragma unroll
    for (int i = 0; i < 4; ++i)
#pragma unroll
      for (int j = 0; j < 2; ++j) acc[i][j] = WMMA_F16(a[i], b[j], acc[i][j]);

    __syncthreads();   // protect LDS tiles before next iteration overwrites
  }

#pragma unroll
  for (int i = 0; i < 4; ++i)
#pragma unroll
    for (int j = 0; j < 2; ++j)
#pragma unroll
      for (int r = 0; r < 8; ++r) {
        const int row = rowBlk + wm * 64 + i * 16 + r + hg * 8;  // M = r + hg*8
        const int col = colBlk + wn * 32 + j * 16 + nl;          // N = lane&15
        const float v = acc[i][j][r];
        if (mode == 0) {
          ((_Float16*)Yout)[(size_t)row * N + col] = (_Float16)v;
        } else if (mode == 1) {                          // V^T store: [b][d][t]
          const int bb = row >> 11, t = row & (Sc - 1);
          ((_Float16*)Yout)[((size_t)bb * N + col) * Sc + t] = (_Float16)v;
        } else if (mode == 2) {                          // sigmoid gate
          const float g = 1.0f / (1.0f + __expf(-(v + bias[col])));
          ((_Float16*)Yout)[(size_t)row * N + col] = (_Float16)g;
        } else {                                         // f32 + bias (final out)
          ((float*)Yout)[(size_t)row * N + col] = v + bias[col];
        }
      }
}

// ---------------------------------------------------------------------------
// Pass 1: per (b,h,s) rowsum[s] = sum_t exp(scale * q_s.k_t).
// Called twice (A/B swapped) to get column sums of the transposed matrix.
// Block = 128 threads = 4 waves; each wave owns 16 rows, loops all t.
// ---------------------------------------------------------------------------
__global__ void attn_rowsum_kernel(const _Float16* __restrict__ A16,
                                   const _Float16* __restrict__ B16,
                                   float* __restrict__ sums) {
  const int lane = threadIdx.x & 31, wid = threadIdx.x >> 5;
  const int hg = lane >> 4, nl = lane & 15;
  const int bh = blockIdx.x, b = bh >> 3, h = bh & 7;
  const int s0 = blockIdx.y * 64 + wid * 16;
  const _Float16* baseA = A16 + (size_t)b * Sc * Dc + h * DKc;
  const _Float16* baseB = B16 + (size_t)b * Sc * Dc + h * DKc;

  v16h a0 = load_a_frag(baseA, Dc, s0, 0, lane);
  v16h a1 = load_a_frag(baseA, Dc, s0, 32, lane);

  float rsum[8];
#pragma unroll
  for (int r = 0; r < 8; ++r) rsum[r] = 0.0f;

  for (int t0 = 0; t0 < Sc; t0 += 16) {
    v8f c = vzero8();
    v16h b0 = load_b_frag(baseB, Dc, t0, 0, lane);
    v16h b1 = load_b_frag(baseB, Dc, t0, 32, lane);
    c = WMMA_F16(a0, b0, c);
    c = WMMA_F16(a1, b1, c);
#pragma unroll
    for (int r = 0; r < 8; ++r) rsum[r] += __expf(c[r] * SCALE);
  }

#pragma unroll
  for (int r = 0; r < 8; ++r) {
    float v = rsum[r];
#pragma unroll
    for (int m = 1; m < 16; m <<= 1) v += __shfl_xor(v, m, 32);
    if (nl == 0) sums[(size_t)bh * Sc + s0 + r + hg * 8] = v;
  }
}

// ---------------------------------------------------------------------------
// Pass 2: recompute score tiles, P = exp(s)^2 * rrow * rcol, O = P @ V,
// gate with sigmoid(xq Wg^T + bg), store gated O (f16) for the out-projection.
// C-fragment -> A-fragment conversion goes through LDS (per-wave tile).
// ---------------------------------------------------------------------------
__global__ void attn_pv_kernel(const _Float16* __restrict__ q16,
                               const _Float16* __restrict__ k16,
                               const _Float16* __restrict__ vt16,
                               const _Float16* __restrict__ gate16,
                               _Float16* __restrict__ go16,
                               const float* __restrict__ rowsum,
                               const float* __restrict__ colsum) {
  __shared__ __align__(16) _Float16 Pt[4][16][40];   // per-wave 16x32 (+pad)
  const int lane = threadIdx.x & 31, wid = threadIdx.x >> 5;
  const int hg = lane >> 4, nl = lane & 15;
  const int bh = blockIdx.x, b = bh >> 3, h = bh & 7;
  const int s0 = blockIdx.y * 64 + wid * 16;

  const _Float16* baseQ  = q16 + (size_t)b * Sc * Dc + h * DKc;
  const _Float16* baseK  = k16 + (size_t)b * Sc * Dc + h * DKc;
  const _Float16* baseVT = vt16 + ((size_t)b * Dc + h * DKc) * Sc;   // [dk][t]
  const float* rs = rowsum + (size_t)bh * Sc;
  const float* cs = colsum + (size_t)bh * Sc;

  float rrow[8];
#pragma unroll
  for (int r = 0; r < 8; ++r) rrow[r] = 1.0f / rs[s0 + r + hg * 8];

  v16h aq0 = load_a_frag(baseQ, Dc, s0, 0, lane);
  v16h aq1 = load_a_frag(baseQ, Dc, s0, 32, lane);

  v8f o[4];
#pragma unroll
  for (int nt = 0; nt < 4; ++nt) o[nt] = vzero8();

  for (int t0 = 0; t0 < Sc; t0 += 32) {
    // ---- scores for two adjacent 16-wide t tiles ----
    v8f c0 = vzero8(), c1 = vzero8();
    v16h bk00 = load_b_frag(baseK, Dc, t0,      0,  lane);
    v16h bk01 = load_b_frag(baseK, Dc, t0,      32, lane);
    v16h bk10 = load_b_frag(baseK, Dc, t0 + 16, 0,  lane);
    v16h bk11 = load_b_frag(baseK, Dc, t0 + 16, 32, lane);
    c0 = WMMA_F16(aq0, bk00, c0);
    c0 = WMMA_F16(aq1, bk01, c0);
    c1 = WMMA_F16(aq0, bk10, c1);
    c1 = WMMA_F16(aq1, bk11, c1);

    const float rc0 = 1.0f / cs[t0 + nl];
    const float rc1 = 1.0f / cs[t0 + 16 + nl];

    // ---- P = exp(s)^2 / (rowsum*colsum), C-frag -> LDS (16x32 f16) ----
#pragma unroll
    for (int r = 0; r < 8; ++r) {
      const float e0 = __expf(c0[r] * SCALE);
      const float e1 = __expf(c1[r] * SCALE);
      Pt[wid][r + hg * 8][nl]      = (_Float16)(e0 * e0 * rrow[r] * rc0);
      Pt[wid][r + hg * 8][nl + 16] = (_Float16)(e1 * e1 * rrow[r] * rc1);
    }
    asm volatile("s_wait_dscnt 0" ::: "memory");   // same-wave LDS RAW

    // ---- LDS -> A fragment (row m = nl, K = hg*16 + 0..15) ----
    const _Float16* pr = &Pt[wid][nl][hg * 16];
    v8h plo = *(const v8h*)pr;
    v8h phi = *(const v8h*)(pr + 8);
    v16h ap;
#pragma unroll
    for (int i = 0; i < 8; ++i) { ap[i] = plo[i]; ap[i + 8] = phi[i]; }

    // ---- O += P @ V  (B operand = contiguous rows of V^T) ----
#pragma unroll
    for (int nt = 0; nt < 4; ++nt) {
      v16h bv = load_b_frag(baseVT, Sc, nt * 16, t0, lane);
      o[nt] = WMMA_F16(ap, bv, o[nt]);
    }
  }

  // ---- gate and store f16 for final projection ----
#pragma unroll
  for (int nt = 0; nt < 4; ++nt)
#pragma unroll
    for (int r = 0; r < 8; ++r) {
      const int srow = s0 + r + hg * 8;
      const int col = h * DKc + nt * 16 + nl;
      const size_t idx = ((size_t)b * Sc + srow) * Dc + col;
      const float g = (float)gate16[idx];
      go16[idx] = (_Float16)(o[nt][r] * g);
    }
}

// ---------------------------------------------------------------------------
// Host launcher
// ---------------------------------------------------------------------------
extern "C" void kernel_launch(void* const* d_in, const int* in_sizes, int n_in,
                              void* d_out, int out_size, void* d_ws, size_t ws_size,
                              hipStream_t stream) {
  (void)in_sizes; (void)n_in; (void)out_size; (void)ws_size;
  const float* x_q   = (const float*)d_in[0];
  const float* x_k   = (const float*)d_in[1];
  const float* Wq    = (const float*)d_in[2];
  const float* Wk    = (const float*)d_in[3];
  const float* Wv    = (const float*)d_in[4];
  const float* Wg    = (const float*)d_in[5];
  const float* bg    = (const float*)d_in[6];
  const float* Wo    = (const float*)d_in[7];
  const float* bo    = (const float*)d_in[8];
  const float* gamma = (const float*)d_in[9];
  const float* beta  = (const float*)d_in[10];
  float* out = (float*)d_out;

  const size_t MS = (size_t)Bc * Sc;        // 8192 rows
  const size_t MD = MS * Dc;                // activations (elems)
  const size_t WW = (size_t)Dc * Dc;        // one weight matrix (elems)

  char* p = (char*)d_ws;
  auto take = [&](size_t bytes) -> char* {
    char* r = p;
    p += (bytes + 255) & ~(size_t)255;
    return r;
  };
  _Float16* xq16   = (_Float16*)take(MD * 2);
  _Float16* xk16   = (_Float16*)take(MD * 2);
  _Float16* q16    = (_Float16*)take(MD * 2);
  _Float16* k16    = (_Float16*)take(MD * 2);
  _Float16* vt16   = (_Float16*)take(MD * 2);
  _Float16* gate16 = (_Float16*)take(MD * 2);
  _Float16* go16   = (_Float16*)take(MD * 2);
  _Float16* Wq16   = (_Float16*)take(WW * 2);
  _Float16* Wk16   = (_Float16*)take(WW * 2);
  _Float16* Wv16   = (_Float16*)take(WW * 2);
  _Float16* Wg16   = (_Float16*)take(WW * 2);
  _Float16* Wo16   = (_Float16*)take(WW * 2);
  float* rowsum = (float*)take((size_t)Bc * Hc * Sc * 4);
  float* colsum = (float*)take((size_t)Bc * Hc * Sc * 4);

  // 1) weights f32 -> f16
  {
    dim3 g((unsigned)((WW + 255) / 256)), t(256);
    f32_to_f16_kernel<<<g, t, 0, stream>>>(Wq, Wq16, (int)WW);
    f32_to_f16_kernel<<<g, t, 0, stream>>>(Wk, Wk16, (int)WW);
    f32_to_f16_kernel<<<g, t, 0, stream>>>(Wv, Wv16, (int)WW);
    f32_to_f16_kernel<<<g, t, 0, stream>>>(Wg, Wg16, (int)WW);
    f32_to_f16_kernel<<<g, t, 0, stream>>>(Wo, Wo16, (int)WW);
  }

  // 2) LayerNorm
  ln_f16_kernel<<<dim3((unsigned)MS), dim3(256), 0, stream>>>(x_q, gamma, beta, xq16);
  ln_f16_kernel<<<dim3((unsigned)MS), dim3(256), 0, stream>>>(x_k, gamma, beta, xk16);

  // 3) projections (WMMA GEMMs, async LDS staging)
  dim3 gg((unsigned)(MS / 128), Dc / 128), tb(256);
  gemm_ws_kernel<<<gg, tb, 0, stream>>>(xq16, Wq16, nullptr, q16,  (int)MS, Dc, Dc, 0);
  gemm_ws_kernel<<<gg, tb, 0, stream>>>(xk16, Wk16, nullptr, k16,  (int)MS, Dc, Dc, 0);
  gemm_ws_kernel<<<gg, tb, 0, stream>>>(xk16, Wv16, nullptr, vt16, (int)MS, Dc, Dc, 1);
  gemm_ws_kernel<<<gg, tb, 0, stream>>>(xq16, Wg16, bg, gate16,    (int)MS, Dc, Dc, 2);

  // 4) softmax denominators (row sums, then column sums via transpose trick)
  dim3 ag(Bc * Hc, Sc / 64), at(128);
  attn_rowsum_kernel<<<ag, at, 0, stream>>>(q16, k16, rowsum);
  attn_rowsum_kernel<<<ag, at, 0, stream>>>(k16, q16, colsum);

  // 5) fused P@V with gating
  attn_pv_kernel<<<ag, at, 0, stream>>>(q16, k16, vt16, gate16, go16, rowsum, colsum);

  // 6) output projection + bias (f32 out)
  gemm_ws_kernel<<<gg, tb, 0, stream>>>(go16, Wo16, bo, out, (int)MS, Dc, Dc, 3);
}
